// FastHybridQuantumAttention_38826504356534
// MI455X (gfx1250) — matrix operand order repaired and checked by reading, not compiled, not run on hardware
//
#include <hip/hip_runtime.h>
#include <hip/hip_bf16.h>
#include <math.h>

// ---------------------------------------------------------------------------
// FastHybridQuantumAttention for gfx1250 (MI455X), wave32 + WMMA bf16.
// Fragments load as pairs of ds_load_b128; contiguous global->LDS tile fills
// use the CDNA5 async-DMA path (GLOBAL_LOAD_ASYNC_TO_LDS_B128 + ASYNCcnt).
// ---------------------------------------------------------------------------

typedef __attribute__((ext_vector_type(16))) __bf16 v16bf;
typedef __attribute__((ext_vector_type(8)))  float  v8f;

union U4 { uint4 q; unsigned short u[8]; };

__device__ __forceinline__ unsigned short f2bf(float f) {
  unsigned int u = __float_as_uint(f);
  u += 0x7fffu + ((u >> 16) & 1u);   // round-to-nearest-even
  return (unsigned short)(u >> 16);
}

// Wave-relative LDS byte offset: low 32 bits of the generic (flat) address
// (LDS aperture maps addr[31:0] directly; HW adds LDS_BASE).
__device__ __forceinline__ unsigned lds_off(const void* p) {
  return (unsigned)(size_t)p;
}
// 16B async copy global -> LDS (GV mode), tracked by ASYNCcnt.
__device__ __forceinline__ void async_copy_b128(unsigned ldsByteOff,
                                                const void* gptr) {
  asm volatile("global_load_async_to_lds_b128 %0, %1, off"
               :: "v"(ldsByteOff), "v"(gptr) : "memory");
}
__device__ __forceinline__ void wait_async0() {
  asm volatile("s_wait_asynccnt 0x0" ::: "memory");
}

// A-fragment (16-bit 16x32 layout): two contiguous 8-elem runs per lane.
__device__ __forceinline__ v16bf load_frag_a(const unsigned short* rowPtr,
                                             int half) {
  union { v16bf v; uint4 q[2]; } f;
  f.q[0] = *(const uint4*)(rowPtr + 8 * half);
  f.q[1] = *(const uint4*)(rowPtr + 16 + 8 * half);
  return f.v;
}
// B-fragment (32x16 layout) from a transposed row (n-major, k contiguous).
__device__ __forceinline__ v16bf load_frag_b(const unsigned short* rowPtr,
                                             int half) {
  union { v16bf v; uint4 q[2]; } f;
  f.q[0] = *(const uint4*)(rowPtr + 8 * half);
  f.q[1] = *(const uint4*)(rowPtr + 16 + 8 * half);
  return f.v;
}

// ---------------------------------------------------------------------------
// fp32 -> bf16 conversion (4-wide)
// ---------------------------------------------------------------------------
__global__ void cvt_f32_bf16(const float* __restrict__ src,
                             unsigned short* __restrict__ dst, int n4) {
  int i = blockIdx.x * blockDim.x + threadIdx.x;
  if (i < n4) {
    float4 f = ((const float4*)src)[i];
    union { unsigned short u[4]; uint2 q; } o;
    o.u[0] = f2bf(f.x); o.u[1] = f2bf(f.y);
    o.u[2] = f2bf(f.z); o.u[3] = f2bf(f.w);
    ((uint2*)dst)[i] = o.q;
  }
}

// ---------------------------------------------------------------------------
// Generic bf16 GEMM: C[M,N] = A[M,K] @ B[K,N]  (row-major, M,N % 64 == 0,
// K % 32 == 0, all % 8). mode 0: bf16 out. mode 1: f32 out + bias.
// Block = 128 threads (4 waves), tile 64x64, wave owns a 16-row band.
// ---------------------------------------------------------------------------
__global__ void __launch_bounds__(128)
gemm_bf16(const unsigned short* __restrict__ A,
          const unsigned short* __restrict__ Bw,
          unsigned short* __restrict__ outb, float* __restrict__ outf,
          const float* __restrict__ bias, int M, int N, int K, int mode) {
  __shared__ __align__(16) unsigned short At[64][32];   // [m][k]
  __shared__ __align__(16) unsigned short BtT[64][40];  // [n][k], padded
  const int tid  = threadIdx.x;
  const int wave = tid >> 5;
  const int lane = tid & 31;
  const int lm   = lane & 15;
  const int half = lane >> 4;
  const int tileM = blockIdx.x * 64;
  const int tileN = blockIdx.y * 64;

  v8f c0 = {}, c1 = {}, c2 = {}, c3 = {};

  for (int k0 = 0; k0 < K; k0 += 32) {
    __syncthreads();
    // A tile: 64 rows x 4 b128 -- async DMA straight into LDS
    for (int e = tid; e < 64 * 4; e += 128) {
      int r = e >> 2, cq = (e & 3) * 8;
      async_copy_b128(lds_off(&At[r][cq]),
                      &A[(size_t)(tileM + r) * K + k0 + cq]);
    }
    // B tile transposed: 32 k-rows x 8 b128 along n, scatter to BtT[n][k]
    for (int e = tid; e < 32 * 8; e += 128) {
      int r = e >> 3, nq = (e & 7) * 8;
      U4 bb; bb.q = *(const uint4*)&Bw[(size_t)(k0 + r) * N + tileN + nq];
#pragma unroll
      for (int t = 0; t < 8; ++t) BtT[nq + t][r] = bb.u[t];
    }
    wait_async0();
    __syncthreads();

    v16bf a = load_frag_a(&At[wave * 16 + lm][0], half);
#pragma unroll
    for (int ns = 0; ns < 4; ++ns) {
      v16bf bv = load_frag_b(&BtT[ns * 16 + lm][0], half);
      v8f& c = ns == 0 ? c0 : ns == 1 ? c1 : ns == 2 ? c2 : c3;
      c = __builtin_amdgcn_wmma_f32_16x16x32_bf16(
            false, a, false, bv, (short)0, c, false, false);
    }
  }

  const int mb = wave * 16 + half * 8;
#pragma unroll
  for (int ns = 0; ns < 4; ++ns) {
    const v8f& c = ns == 0 ? c0 : ns == 1 ? c1 : ns == 2 ? c2 : c3;
    int col = tileN + ns * 16 + lm;
#pragma unroll
    for (int r = 0; r < 8; ++r) {
      int row = tileM + mb + r;
      float val = c[r];
      if (mode == 1) outf[(size_t)row * N + col] = val + bias[col];
      else           outb[(size_t)row * N + col] = f2bf(val);
    }
  }
}

// ---------------------------------------------------------------------------
// Flash attention for traditional heads. One block = (b, head, 16-query tile).
// 128 threads = 4 waves; wave w owns output columns [16w, 16w+16).
// Kt[key][d] is directly the transposed layout needed for S = Q K^T;
// V is stored transposed (VtT[d][key]) for the P@V WMMA.
// Q and K tiles are filled via async DMA; V needs an on-the-fly transpose.
// ---------------------------------------------------------------------------
__global__ void __launch_bounds__(128)
flash_attn(const unsigned short* __restrict__ Q,
           const unsigned short* __restrict__ Kb,
           const unsigned short* __restrict__ Vb,
           unsigned short* __restrict__ cat,
           int Bsz, int Lsz, int THn) {
  const int nqt = Lsz / 16;
  const int blk = blockIdx.x;
  const int qt  = blk % nqt;
  const int h   = (blk / nqt) % THn;
  const int b   = blk / (nqt * THn);
  const int stride = THn * 64;

  __shared__ __align__(16) unsigned short Qt[16][64];   // [q][d]
  __shared__ __align__(16) unsigned short Kt[64][64];   // [key][d]
  __shared__ __align__(16) unsigned short VtT[64][72];  // [d][key], padded
  __shared__ float          St[16][64];
  __shared__ __align__(16) unsigned short Pt[16][64];   // [q][key]
  __shared__ float m_s[16], l_s[16], alpha_s[16];

  const int tid  = threadIdx.x;
  const int wave = tid >> 5;
  const int lane = tid & 31;
  const int lm   = lane & 15;
  const int half = lane >> 4;

  // Q tile: 16 rows x 8 b128, one async copy per thread
  {
    int r = tid >> 3, dq = (tid & 7) * 8;
    async_copy_b128(lds_off(&Qt[r][dq]),
                    &Q[(size_t)(b * Lsz + qt * 16 + r) * stride + h * 64 + dq]);
  }
  if (tid < 16) { m_s[tid] = -__builtin_inff(); l_s[tid] = 0.f; }

  v8f o = {};
  const float scale = 0.125f;  // 1/sqrt(64)

  for (int kb = 0; kb < Lsz / 64; ++kb) {
    __syncthreads();
    for (int e = tid; e < 64 * 8; e += 128) {
      int r = e >> 3, dq = (e & 7) * 8;
      size_t base = (size_t)(b * Lsz + kb * 64 + r) * stride + h * 64 + dq;
      async_copy_b128(lds_off(&Kt[r][dq]), &Kb[base]);     // K: direct DMA
      U4 vv; vv.q = *(const uint4*)&Vb[base];              // V: transpose
#pragma unroll
      for (int t = 0; t < 8; ++t) VtT[dq + t][r] = vv.u[t];
    }
    wait_async0();
    __syncthreads();

    // ---- S = Q @ K^T (wave w -> keys [16w,16w+16)) ----
    v8f cs = {};
#pragma unroll
    for (int k0 = 0; k0 < 64; k0 += 32) {
      v16bf a  = load_frag_a(&Qt[lm][k0], half);
      v16bf bv = load_frag_b(&Kt[wave * 16 + lm][k0], half);
      cs = __builtin_amdgcn_wmma_f32_16x16x32_bf16(
             false, a, false, bv, (short)0, cs, false, false);
    }
#pragma unroll
    for (int r = 0; r < 8; ++r)
      St[half * 8 + r][wave * 16 + lm] = cs[r] * scale;
    __syncthreads();

    // ---- online softmax (one thread per query row) ----
    if (tid < 16) {
      float mo = m_s[tid];
      float mx = mo;
      for (int j = 0; j < 64; ++j) mx = fmaxf(mx, St[tid][j]);
      float al = expf(mo - mx);
      float lsum = 0.f;
      for (int j = 0; j < 64; ++j) {
        float p = expf(St[tid][j] - mx);
        lsum += p;
        Pt[tid][j] = f2bf(p);
      }
      m_s[tid] = mx;
      l_s[tid] = l_s[tid] * al + lsum;
      alpha_s[tid] = al;
    }
    __syncthreads();

    // ---- O = O*alpha + P @ V ----
#pragma unroll
    for (int r = 0; r < 8; ++r) o[r] *= alpha_s[half * 8 + r];
#pragma unroll
    for (int k0 = 0; k0 < 64; k0 += 32) {
      v16bf a  = load_frag_a(&Pt[lm][k0], half);
      v16bf bv = load_frag_b(&VtT[wave * 16 + lm][k0], half);
      o = __builtin_amdgcn_wmma_f32_16x16x32_bf16(
            false, a, false, bv, (short)0, o, false, false);
    }
  }
  __syncthreads();
#pragma unroll
  for (int r = 0; r < 8; ++r) {
    int row = half * 8 + r;
    float val = o[r] / l_s[row];
    int lg = qt * 16 + row;
    cat[(size_t)(b * Lsz + lg) * 1024 + 128 + h * 64 + wave * 16 + lm] = f2bf(val);
  }
}

// ---------------------------------------------------------------------------
// Quantum state: h = xh @ wq1 + bq1; LayerNorm(128); tanh; split amp/phase.
// One 128-thread block per (b, l, head).
// ---------------------------------------------------------------------------
__global__ void __launch_bounds__(128)
quantum_state(const float* __restrict__ x, const float* __restrict__ wq1,
              const float* __restrict__ bq1, const float* __restrict__ ln_g,
              const float* __restrict__ ln_b, float* __restrict__ amp,
              float* __restrict__ phase, int Bsz, int Lsz) {
  const int idx  = blockIdx.x;
  const int head = idx & 1;
  const int l    = (idx >> 1) % Lsz;
  const int b    = idx / (2 * Lsz);
  __shared__ float xh[64];
  __shared__ float red[128];
  const int j = threadIdx.x;
  if (j < 64) xh[j] = x[(size_t)(b * Lsz + l) * 1024 + head * 64 + j];
  __syncthreads();
  float hj = bq1[j];
  for (int k = 0; k < 64; ++k) hj += xh[k] * wq1[k * 128 + j];
  red[j] = hj; __syncthreads();
  for (int off = 64; off > 0; off >>= 1) {
    if (j < off) red[j] += red[j + off];
    __syncthreads();
  }
  float mu = red[0] * (1.0f / 128.0f);
  __syncthreads();
  float d = hj - mu;
  red[j] = d * d; __syncthreads();
  for (int off = 64; off > 0; off >>= 1) {
    if (j < off) red[j] += red[j + off];
    __syncthreads();
  }
  float var = red[0] * (1.0f / 128.0f);
  float t = tanhf(d * rsqrtf(var + 1e-5f) * ln_g[j] + ln_b[j]);
  size_t base = ((size_t)(head * Bsz + b) * Lsz + l) * 64;
  if (j < 64) amp[base + j] = t;
  else        phase[base + (j - 64)] = t;
}

// ---------------------------------------------------------------------------
// Quantum chunk attention. One 64-thread block per (b, head, chunk); thread i
// handles query row i. Keys = positions 0..63. cos-table lookup replicated as
// cos(idx * 2pi/2047), idx = floor(mod(pd/2pi*2048, 2048)).
// ---------------------------------------------------------------------------
__global__ void __launch_bounds__(64)
quantum_attn(const float* __restrict__ x, const float* __restrict__ amp,
             const float* __restrict__ phase, const float* __restrict__ freqs,
             const float* __restrict__ shifts, unsigned short* __restrict__ cat,
             int Bsz, int Lsz) {
  const int nc   = Lsz / 64;
  const int blk  = blockIdx.x;
  const int c    = blk % nc;
  const int head = (blk / nc) & 1;
  const int b    = blk / (nc * 2);
  __shared__ float ka[64][64];
  __shared__ float kp[64][64];
  __shared__ float vt[64][64];
  const int tid = threadIdx.x;
  const size_t abase = ((size_t)(head * Bsz + b) * Lsz) * 64;
  for (int e = tid; e < 64 * 64; e += 64) {
    int r = e >> 6, d0 = e & 63;
    ka[r][d0] = amp[abase + (size_t)r * 64 + d0];
    kp[r][d0] = phase[abase + (size_t)r * 64 + d0];
    vt[r][d0] = x[(size_t)(b * Lsz + r) * 1024 + head * 64 + d0];
  }
  __syncthreads();
  const int i = tid;
  const size_t qbase = abase + (size_t)(c * 64 + i) * 64;
  float qa[64], qp[64], s[64];
  for (int d0 = 0; d0 < 64; ++d0) {
    qa[d0] = amp[qbase + d0];
    qp[d0] = phase[qbase + d0];
  }
  const float fr = freqs[head], sh = shifts[head];
  const float TWO_PI = 6.283185307179586f;
  float mx = -__builtin_inff();
  for (int jj = 0; jj < 64; ++jj) {
    float dot = 0.f;
    for (int d0 = 0; d0 < 64; ++d0) dot += qa[d0] * ka[jj][d0];
    float pd = (qp[jj] - kp[jj][i]) * fr + sh;
    float t = pd * (2048.0f / TWO_PI);
    float r = fmodf(t, 2048.0f);
    if (r < 0.f) r += 2048.0f;
    int idxq = (int)r;
    if (idxq > 2047) idxq = 2047;
    float interf = __cosf((float)idxq * (TWO_PI / 2047.0f));
    s[jj] = dot * interf * 0.125f;
    mx = fmaxf(mx, s[jj]);
  }
  float lsum = 0.f;
  for (int jj = 0; jj < 64; ++jj) { s[jj] = expf(s[jj] - mx); lsum += s[jj]; }
  const float inv = 1.0f / lsum;
  const size_t obase = (size_t)(b * Lsz + c * 64 + i) * 1024 + head * 64;
  for (int d0 = 0; d0 < 64; ++d0) {
    float acc = 0.f;
    for (int jj = 0; jj < 64; ++jj) acc += s[jj] * vt[jj][d0];
    cat[obase + d0] = f2bf(acc * inv);
  }
}

// ---------------------------------------------------------------------------
// Host orchestration
// ---------------------------------------------------------------------------
extern "C" void kernel_launch(void* const* d_in, const int* in_sizes, int n_in,
                              void* d_out, int out_size, void* d_ws,
                              size_t ws_size, hipStream_t stream) {
  (void)in_sizes; (void)n_in; (void)out_size; (void)ws_size;
  const float* x      = (const float*)d_in[0];
  const float* wq1    = (const float*)d_in[1];
  const float* bq1    = (const float*)d_in[2];
  const float* ln_g   = (const float*)d_in[3];
  const float* ln_b   = (const float*)d_in[4];
  const float* shifts = (const float*)d_in[5];
  const float* freqs  = (const float*)d_in[6];
  const float* wq     = (const float*)d_in[7];
  const float* wk     = (const float*)d_in[8];
  const float* wv     = (const float*)d_in[9];
  const float* wo     = (const float*)d_in[10];
  const float* bo     = (const float*)d_in[11];
  float* out = (float*)d_out;

  const int Bz = 2, Lz = 2048, DIMz = 1024, THz = 14;
  const int M  = Bz * Lz;      // 4096
  const int Np = THz * 64;     // 896

  char* ws = (char*)d_ws;
  size_t off = 0;
  auto alloc = [&](size_t bytes) -> void* {
    void* p = ws + off;
    off += (bytes + 255) & ~(size_t)255;
    return p;
  };
  unsigned short* x_bf   = (unsigned short*)alloc((size_t)M * DIMz * 2);
  unsigned short* wq_bf  = (unsigned short*)alloc((size_t)DIMz * Np * 2);
  unsigned short* wk_bf  = (unsigned short*)alloc((size_t)DIMz * Np * 2);
  unsigned short* wv_bf  = (unsigned short*)alloc((size_t)DIMz * Np * 2);
  unsigned short* wo_bf  = (unsigned short*)alloc((size_t)DIMz * DIMz * 2);
  unsigned short* q_bf   = (unsigned short*)alloc((size_t)M * Np * 2);
  unsigned short* k_bf   = (unsigned short*)alloc((size_t)M * Np * 2);
  unsigned short* v_bf   = (unsigned short*)alloc((size_t)M * Np * 2);
  unsigned short* cat_bf = (unsigned short*)alloc((size_t)M * DIMz * 2);
  float* amp_f   = (float*)alloc((size_t)2 * Bz * Lz * 64 * 4);
  float* phase_f = (float*)alloc((size_t)2 * Bz * Lz * 64 * 4);

  auto cvt = [&](const float* s, unsigned short* d, int n) {
    int n4 = n / 4;
    cvt_f32_bf16<<<(n4 + 255) / 256, 256, 0, stream>>>(s, d, n4);
  };
  cvt(x,  x_bf,  M * DIMz);
  cvt(wq, wq_bf, DIMz * Np);
  cvt(wk, wk_bf, DIMz * Np);
  cvt(wv, wv_bf, DIMz * Np);
  cvt(wo, wo_bf, DIMz * DIMz);

  // Q/K/V projections (bf16 out)
  dim3 gp(M / 64, Np / 64);
  gemm_bf16<<<gp, 128, 0, stream>>>(x_bf, wq_bf, q_bf, nullptr, nullptr,
                                    M, Np, DIMz, 0);
  gemm_bf16<<<gp, 128, 0, stream>>>(x_bf, wk_bf, k_bf, nullptr, nullptr,
                                    M, Np, DIMz, 0);
  gemm_bf16<<<gp, 128, 0, stream>>>(x_bf, wv_bf, v_bf, nullptr, nullptr,
                                    M, Np, DIMz, 0);

  // Quantum heads -> cat columns [0,128)
  quantum_state<<<Bz * Lz * 2, 128, 0, stream>>>(x, wq1, bq1, ln_g, ln_b,
                                                 amp_f, phase_f, Bz, Lz);
  quantum_attn<<<Bz * 2 * (Lz / 64), 64, 0, stream>>>(x, amp_f, phase_f, freqs,
                                                      shifts, cat_bf, Bz, Lz);

  // Traditional heads (flash) -> cat columns [128,1024)
  flash_attn<<<Bz * THz * (Lz / 16), 128, 0, stream>>>(q_bf, k_bf, v_bf,
                                                       cat_bf, Bz, Lz, THz);

  // Output projection: out = cat @ wo + bo (f32)
  dim3 go(M / 64, DIMz / 64);
  gemm_bf16<<<go, 128, 0, stream>>>(cat_bf, wo_bf, nullptr, out, bo,
                                    M, DIMz, DIMz, 1);
}
